// VQAShotSelector_62758062129675
// MI455X (gfx1250) — compile-verified, matrix operand order checked by reading, not run on hardware
//
#include <hip/hip_runtime.h>
#include <math.h>

// ---------------------------------------------------------------------------
// Types for CDNA5 WMMA (gfx1250, wave32)
// ---------------------------------------------------------------------------
typedef __attribute__((ext_vector_type(16))) __bf16 v16bf;
typedef __attribute__((ext_vector_type(8)))  float  v8f;
typedef __attribute__((ext_vector_type(4)))  unsigned u32x4;
typedef __attribute__((ext_vector_type(4)))  int      i32x4;
typedef __attribute__((ext_vector_type(8)))  int      i32x8;

__device__ __forceinline__ v8f wmma_bf16(v16bf a, v16bf b, v8f c) {
  // (neg_a, A, neg_b, B, c_mod, C, reuse_a, reuse_b)
  return __builtin_amdgcn_wmma_f32_16x16x32_bf16(false, a, false, b,
                                                 (short)0, c, false, false);
}

#if defined(__has_builtin)
#if __has_builtin(__builtin_amdgcn_tensor_load_to_lds) && \
    __has_builtin(__builtin_amdgcn_s_wait_tensorcnt)
#define USE_TDM 1
#endif
#endif
#ifndef USE_TDM
#define USE_TDM 0
#endif

#define S_ 512
#define B_ 64
#define D_ 1024
#define H_ 8
#define DH_ 128
#define DFF_ 2048
#define L_ 6
#define NTOK (S_ * B_)

// ---------------------------------------------------------------------------
// Positional encoding add: x[s,b,d] = src[s,b,d] + PE(s,d)
// ---------------------------------------------------------------------------
__global__ __launch_bounds__(256) void posenc_kernel(
    const float* __restrict__ src, float* __restrict__ x)
{
  size_t idx = (size_t)blockIdx.x * 256 + threadIdx.x;
  if (idx >= (size_t)NTOK * D_) return;
  int d = (int)(idx % D_);
  size_t sb = idx / D_;
  int s = (int)(sb / B_);
  int i2 = d & ~1;
  float freq = __expf(-(float)i2 * 0.008994473310147239f); // ln(10000)/1024
  float ang = (float)s * freq;
  float pe = (d & 1) ? __cosf(ang) : __sinf(ang);
  x[idx] = src[idx] + pe;
}

// ---------------------------------------------------------------------------
// Generic GEMM: Y[N,M] = X[N,K] @ W[M,K]^T + bias (+residual, optional ReLU)
// fp32 in/out, bf16 WMMA compute. N,M multiples of 64; K multiple of 32.
// Block = 256 threads (8 waves). 64x64 output tile, K-step 32.
// ---------------------------------------------------------------------------
#define GEMM_LDK 36   // 32 + 4 pad

// ---- fallback path: VGPR-staged LDS tiles (bf16) --------------------------
__global__ __launch_bounds__(256) void gemm_bf16_kernel(
    const float* __restrict__ X, const float* __restrict__ W,
    const float* __restrict__ bias, const float* __restrict__ residual,
    float* __restrict__ Y, int N, int K, int M, int do_relu)
{
  __shared__ __bf16 Xs[64][GEMM_LDK];
  __shared__ __bf16 Ws[64][GEMM_LDK];

  const int tid  = threadIdx.x;
  const int lane = tid & 31;
  const int wave = tid >> 5;

  const int bm = blockIdx.x * 64;
  const int bn = blockIdx.y * 64;

  const int r  = wave & 3;
  const int c0 = (wave >> 2) * 2;

  const int lrow = tid >> 2;
  const int lcol = (tid & 3) * 8;

  const int arow   = r * 16 + (lane & 15);
  const int koff   = (lane >> 4) * 16;
  const int col    = lane & 15;
  const int rowoff = (lane >> 4) * 8;

  v8f acc0 = {};
  v8f acc1 = {};

  for (int k0 = 0; k0 < K; k0 += 32) {
    const float* xg = X + (size_t)(bm + lrow) * K + k0 + lcol;
    const float* wg = W + (size_t)(bn + lrow) * K + k0 + lcol;
    float4 x0 = ((const float4*)xg)[0];
    float4 x1 = ((const float4*)xg)[1];
    float4 w0 = ((const float4*)wg)[0];
    float4 w1 = ((const float4*)wg)[1];

    __syncthreads();
    Xs[lrow][lcol + 0] = (__bf16)x0.x; Xs[lrow][lcol + 1] = (__bf16)x0.y;
    Xs[lrow][lcol + 2] = (__bf16)x0.z; Xs[lrow][lcol + 3] = (__bf16)x0.w;
    Xs[lrow][lcol + 4] = (__bf16)x1.x; Xs[lrow][lcol + 5] = (__bf16)x1.y;
    Xs[lrow][lcol + 6] = (__bf16)x1.z; Xs[lrow][lcol + 7] = (__bf16)x1.w;
    Ws[lrow][lcol + 0] = (__bf16)w0.x; Ws[lrow][lcol + 1] = (__bf16)w0.y;
    Ws[lrow][lcol + 2] = (__bf16)w0.z; Ws[lrow][lcol + 3] = (__bf16)w0.w;
    Ws[lrow][lcol + 4] = (__bf16)w1.x; Ws[lrow][lcol + 5] = (__bf16)w1.y;
    Ws[lrow][lcol + 6] = (__bf16)w1.z; Ws[lrow][lcol + 7] = (__bf16)w1.w;
    __syncthreads();

    v16bf a, b0, b1;
#pragma unroll
    for (int i = 0; i < 16; ++i) {
      a[i]  = Xs[arow][koff + i];
      b0[i] = Ws[c0 * 16 + col][koff + i];
      b1[i] = Ws[(c0 + 1) * 16 + col][koff + i];
    }
    acc0 = wmma_bf16(a, b0, acc0);
    acc1 = wmma_bf16(a, b1, acc1);
  }

#pragma unroll
  for (int j = 0; j < 8; ++j) {
    int gr  = bm + r * 16 + j + rowoff;
    int gc0 = bn + c0 * 16 + col;
    int gc1 = gc0 + 16;
    float v0 = acc0[j] + bias[gc0];
    float v1 = acc1[j] + bias[gc1];
    if (do_relu) { v0 = fmaxf(v0, 0.f); v1 = fmaxf(v1, 0.f); }
    if (residual) {
      v0 += residual[(size_t)gr * M + gc0];
      v1 += residual[(size_t)gr * M + gc1];
    }
    Y[(size_t)gr * M + gc0] = v0;
    Y[(size_t)gr * M + gc1] = v1;
  }
}

#if USE_TDM
// ---------------------------------------------------------------------------
// TDM helper: DMA a 64-row x 32-col fp32 tile (row stride `ld` elems) from
// global memory into LDS at byte offset lds_off, with D# LDS padding of
// 4 DWORDs every 32 DWORDs -> padded row stride of 36 floats (GEMM_LDK).
// D# packing per CDNA5 ISA sec. 8.3 / 8.4.
// ---------------------------------------------------------------------------
__device__ __forceinline__ void tdm_load_tile64x32(unsigned lds_off,
                                                   const float* gsrc, int ld)
{
  unsigned long long ga = (unsigned long long)(const void*)gsrc;
  u32x4 g0 = {};
  g0[0] = 1u;                                   // count=1, user descriptor
  g0[1] = lds_off;                              // LDS byte address
  g0[2] = (unsigned)ga;                         // global_addr[31:0]
  g0[3] = (unsigned)((ga >> 32) & 0x01FFFFFFu)  // global_addr[56:32]
        | 0x80000000u;                          // type=2 ("image") in [127:126]

  unsigned dim0 = (unsigned)ld;                 // tensor_dim0 (elements)
  unsigned dim1 = 1u << 20;                     // tensor_dim1 (generous)
  unsigned strd = (unsigned)ld;                 // tensor_dim0_stride (elements)

  i32x8 g1 = {};
  g1[0] = (2 << 16)    // data_size = 4 bytes
        | (1 << 20)    // pad_enable
        | (4 << 22)    // pad_interval: 2^(4+1)=32 DWORDs
        | (3 << 25);   // pad_amount: 3+1 = 4 DWORDs
  g1[1] = (int)((dim0 & 0xFFFFu) << 16);                    // dim0[15:0]
  g1[2] = (int)((dim0 >> 16) | ((dim1 & 0xFFFFu) << 16));   // dim0[31:16]|dim1[15:0]
  g1[3] = (int)((dim1 >> 16) | (32u << 16));                // dim1[31:16]|tile_dim0=32
  g1[4] = (int)64;                                          // tile_dim1=64, tile_dim2=0
  g1[5] = (int)strd;                                        // dim0_stride[31:0]
  g1[6] = 0;                                                // stride[47:32], dim1_stride lo
  g1[7] = 0;

  i32x4 z4 = {};
#if __clang_major__ >= 23
  i32x8 z8 = {};
  __builtin_amdgcn_tensor_load_to_lds(g0, g1, z4, z4, z8, 0);
#else
  __builtin_amdgcn_tensor_load_to_lds(g0, g1, z4, z4, 0);
#endif
}

__device__ __forceinline__ unsigned lds_byte_off(const void* p)
{
  // Flat LDS aperture: addr[31:0] is the wave-relative LDS byte offset.
  return (unsigned)(unsigned long long)p;
}

// ---- TDM path: double-buffered fp32 tiles DMA'd into LDS ------------------
__global__ __launch_bounds__(256) void gemm_tdm_kernel(
    const float* __restrict__ X, const float* __restrict__ W,
    const float* __restrict__ bias, const float* __restrict__ residual,
    float* __restrict__ Y, int N, int K, int M, int do_relu)
{
  __shared__ float XsF[2][64][GEMM_LDK];   // 2 x 9216 B
  __shared__ float WsF[2][64][GEMM_LDK];   // 2 x 9216 B

  const int tid  = threadIdx.x;
  const int lane = tid & 31;
  const int wave = tid >> 5;

  const int bm = blockIdx.x * 64;
  const int bn = blockIdx.y * 64;

  const int r  = wave & 3;
  const int c0 = (wave >> 2) * 2;

  const int arow   = r * 16 + (lane & 15);
  const int koff   = (lane >> 4) * 16;
  const int col    = lane & 15;
  const int rowoff = (lane >> 4) * 8;

  v8f acc0 = {};
  v8f acc1 = {};

  const float* Xrow = X + (size_t)bm * K;
  const float* Wrow = W + (size_t)bn * K;

  if (wave == 0) {
    tdm_load_tile64x32(lds_byte_off(&XsF[0][0][0]), Xrow, K);
    tdm_load_tile64x32(lds_byte_off(&WsF[0][0][0]), Wrow, K);
  }

  int buf = 0;
  for (int k0 = 0; k0 < K; k0 += 32, buf ^= 1) {
    if (wave == 0) {
      __builtin_amdgcn_s_wait_tensorcnt(0);   // current buffer's DMAs done
    }
    __syncthreads();                          // publish to all waves
    if (wave == 0 && (k0 + 32) < K) {
      // consumers of buf^1 passed the barrier above -> safe to overwrite
      tdm_load_tile64x32(lds_byte_off(&XsF[buf ^ 1][0][0]), Xrow + k0 + 32, K);
      tdm_load_tile64x32(lds_byte_off(&WsF[buf ^ 1][0][0]), Wrow + k0 + 32, K);
    }

    v16bf a, b0, b1;
#pragma unroll
    for (int i = 0; i < 16; ++i) {
      a[i]  = (__bf16)XsF[buf][arow][koff + i];
      b0[i] = (__bf16)WsF[buf][c0 * 16 + col][koff + i];
      b1[i] = (__bf16)WsF[buf][(c0 + 1) * 16 + col][koff + i];
    }
    acc0 = wmma_bf16(a, b0, acc0);
    acc1 = wmma_bf16(a, b1, acc1);
  }

#pragma unroll
  for (int j = 0; j < 8; ++j) {
    int gr  = bm + r * 16 + j + rowoff;
    int gc0 = bn + c0 * 16 + col;
    int gc1 = gc0 + 16;
    float v0 = acc0[j] + bias[gc0];
    float v1 = acc1[j] + bias[gc1];
    if (do_relu) { v0 = fmaxf(v0, 0.f); v1 = fmaxf(v1, 0.f); }
    if (residual) {
      v0 += residual[(size_t)gr * M + gc0];
      v1 += residual[(size_t)gr * M + gc1];
    }
    Y[(size_t)gr * M + gc0] = v0;
    Y[(size_t)gr * M + gc1] = v1;
  }
}
#endif  // USE_TDM

// ---------------------------------------------------------------------------
// Flash attention (per (b,h), 128 query rows per block, 8 waves x 16 rows).
// qkv layout: [S*B, 3*D], token t = s*B + b; q at +0, k at +D, v at +2D.
// ---------------------------------------------------------------------------
__global__ __launch_bounds__(256) void flash_attn_kernel(
    const float* __restrict__ qkv, float* __restrict__ out)
{
  __shared__ __bf16 Qs[128][DH_ + 4];
  __shared__ __bf16 Ks[32][DH_ + 4];
  __shared__ __bf16 Vt[DH_][36];      // V transposed: [d][key]
  __shared__ __bf16 Ps[8][16][36];    // per-wave P tile

  const int tid  = threadIdx.x;
  const int lane = tid & 31;
  const int wave = tid >> 5;

  const int qb = blockIdx.x & 3;
  const int bh = blockIdx.x >> 2;
  const int b  = bh >> 3;
  const int h  = bh & 7;
  const int qbase = qb * 128;
  const int D3 = 3 * D_;
  const float scale = 0.08838834764831845f;   // 1/sqrt(128)

  {
    int row = tid >> 1;
    int c0  = (tid & 1) * 64;
    const float* qg = qkv + (size_t)((qbase + row) * B_ + b) * D3 + h * DH_ + c0;
#pragma unroll
    for (int i = 0; i < 64; i += 4) {
      float4 v = *(const float4*)(qg + i);
      Qs[row][c0 + i + 0] = (__bf16)(v.x * scale);
      Qs[row][c0 + i + 1] = (__bf16)(v.y * scale);
      Qs[row][c0 + i + 2] = (__bf16)(v.z * scale);
      Qs[row][c0 + i + 3] = (__bf16)(v.w * scale);
    }
  }

  const int arow   = lane & 15;
  const int koff   = (lane >> 4) * 16;
  const int col    = lane & 15;
  const int rowoff = (lane >> 4) * 8;

  float Mrow[8], Lrow[8];
  v8f Oacc[8];
#pragma unroll
  for (int j = 0; j < 8; ++j) { Mrow[j] = -1e30f; Lrow[j] = 0.f; }
#pragma unroll
  for (int d = 0; d < 8; ++d) { v8f z = {}; Oacc[d] = z; }

  for (int kt = 0; kt < S_; kt += 32) {
    __syncthreads();
    {
      int row = tid >> 3;
      int c0  = (tid & 7) * 16;
      const float* kg = qkv + (size_t)((kt + row) * B_ + b) * D3 + D_ + h * DH_ + c0;
      const float* vg = kg + D_;
#pragma unroll
      for (int i = 0; i < 16; i += 4) {
        float4 kv = *(const float4*)(kg + i);
        float4 vv = *(const float4*)(vg + i);
        Ks[row][c0 + i + 0] = (__bf16)kv.x;
        Ks[row][c0 + i + 1] = (__bf16)kv.y;
        Ks[row][c0 + i + 2] = (__bf16)kv.z;
        Ks[row][c0 + i + 3] = (__bf16)kv.w;
        Vt[c0 + i + 0][row] = (__bf16)vv.x;
        Vt[c0 + i + 1][row] = (__bf16)vv.y;
        Vt[c0 + i + 2][row] = (__bf16)vv.z;
        Vt[c0 + i + 3][row] = (__bf16)vv.w;
      }
    }
    __syncthreads();

    v8f s0 = {}, s1 = {};
#pragma unroll
    for (int kk = 0; kk < DH_; kk += 32) {
      v16bf a, b0, b1;
#pragma unroll
      for (int i = 0; i < 16; ++i) {
        a[i]  = Qs[wave * 16 + arow][kk + koff + i];
        b0[i] = Ks[col][kk + koff + i];
        b1[i] = Ks[16 + col][kk + koff + i];
      }
      s0 = wmma_bf16(a, b0, s0);
      s1 = wmma_bf16(a, b1, s1);
    }

#pragma unroll
    for (int j = 0; j < 8; ++j) {
      float v0 = s0[j], v1 = s1[j];
      float mx = fmaxf(v0, v1);
      mx = fmaxf(mx, __shfl_xor(mx, 1));
      mx = fmaxf(mx, __shfl_xor(mx, 2));
      mx = fmaxf(mx, __shfl_xor(mx, 4));
      mx = fmaxf(mx, __shfl_xor(mx, 8));
      float mold = Mrow[j];
      float mnew = fmaxf(mold, mx);
      float corr = __expf(mold - mnew);
      float p0 = __expf(v0 - mnew);
      float p1 = __expf(v1 - mnew);
      float rs = p0 + p1;
      rs += __shfl_xor(rs, 1);
      rs += __shfl_xor(rs, 2);
      rs += __shfl_xor(rs, 4);
      rs += __shfl_xor(rs, 8);
      Lrow[j] = Lrow[j] * corr + rs;
      Mrow[j] = mnew;
#pragma unroll
      for (int d = 0; d < 8; ++d) Oacc[d][j] *= corr;
      Ps[wave][j + rowoff][col]      = (__bf16)p0;
      Ps[wave][j + rowoff][16 + col] = (__bf16)p1;
    }
    asm volatile("" ::: "memory");   // order P stores before P loads (same wave)

    v16bf pa;
#pragma unroll
    for (int i = 0; i < 16; ++i) pa[i] = Ps[wave][arow][koff + i];
#pragma unroll
    for (int d = 0; d < 8; ++d) {
      v16bf bv;
#pragma unroll
      for (int i = 0; i < 16; ++i) bv[i] = Vt[d * 16 + col][koff + i];
      Oacc[d] = wmma_bf16(pa, bv, Oacc[d]);
    }
  }

#pragma unroll
  for (int j = 0; j < 8; ++j) {
    float inv = 1.0f / Lrow[j];
    int srow = qbase + wave * 16 + j + rowoff;
#pragma unroll
    for (int d = 0; d < 8; ++d) {
      out[(size_t)(srow * B_ + b) * D_ + h * DH_ + d * 16 + col] = Oacc[d][j] * inv;
    }
  }
}

// ---------------------------------------------------------------------------
// LayerNorm: one block per token row of D_ elements.
// ---------------------------------------------------------------------------
__global__ __launch_bounds__(256) void layernorm_kernel(
    const float* __restrict__ in, const float* __restrict__ g,
    const float* __restrict__ be, float* __restrict__ out)
{
  __shared__ float ws1[8], ws2[8];
  int t = blockIdx.x;
  const float* x = in + (size_t)t * D_;
  float s = 0.f, s2 = 0.f;
  for (int i = threadIdx.x; i < D_; i += 256) {
    float v = x[i]; s += v; s2 += v * v;
  }
#pragma unroll
  for (int off = 16; off; off >>= 1) {
    s  += __shfl_xor(s, off);
    s2 += __shfl_xor(s2, off);
  }
  if ((threadIdx.x & 31) == 0) { ws1[threadIdx.x >> 5] = s; ws2[threadIdx.x >> 5] = s2; }
  __syncthreads();
  float ts = 0.f, ts2 = 0.f;
#pragma unroll
  for (int i = 0; i < 8; ++i) { ts += ws1[i]; ts2 += ws2[i]; }
  float mean = ts * (1.0f / D_);
  float var  = ts2 * (1.0f / D_) - mean * mean;
  float rstd = rsqrtf(var + 1e-5f);
  for (int i = threadIdx.x; i < D_; i += 256) {
    out[(size_t)t * D_ + i] = (x[i] - mean) * rstd * g[i] + be[i];
  }
}

// ---------------------------------------------------------------------------
// Decoder logits: logits[(b*H+h)*S + s] = dot(q[b,h,:], k[s,b,h,:]) / sqrt(DH)
// ---------------------------------------------------------------------------
__global__ __launch_bounds__(256) void dec_logits_kernel(
    const float* __restrict__ q, const float* __restrict__ k,
    float* __restrict__ logits)
{
  int idx = blockIdx.x * 256 + threadIdx.x;
  if (idx >= B_ * H_ * S_) return;
  int s  = idx % S_;
  int bh = idx / S_;
  int h  = bh % H_;
  int b  = bh / H_;
  const float* qp = q + (size_t)b * D_ + h * DH_;
  const float* kp = k + (size_t)(s * B_ + b) * D_ + h * DH_;
  float acc = 0.f;
#pragma unroll 4
  for (int d = 0; d < DH_; ++d) acc += qp[d] * kp[d];
  logits[(size_t)(b * H_ + h) * S_ + s] = acc * 0.08838834764831845f;
}

// ---------------------------------------------------------------------------
// Pointer-decoder sampling. One block per batch row.
// ---------------------------------------------------------------------------
__device__ __forceinline__ float block_reduce(float v, float* buf, bool ismax)
{
#pragma unroll
  for (int off = 16; off; off >>= 1) {
    float o = __shfl_xor(v, off);
    v = ismax ? fmaxf(v, o) : (v + o);
  }
  if ((threadIdx.x & 31) == 0) buf[threadIdx.x >> 5] = v;
  __syncthreads();
  float r = buf[0];
#pragma unroll
  for (int i = 1; i < 8; ++i) r = ismax ? fmaxf(r, buf[i]) : (r + buf[i]);
  __syncthreads();
  return r;
}

__global__ __launch_bounds__(256) void sampler_kernel(
    const float* __restrict__ logits, float* __restrict__ outIdx,
    float* __restrict__ outProbs, int steps)
{
  __shared__ float sc[S_];
  __shared__ float wred[8];
  __shared__ unsigned msk[S_ / 32];
  const int b = blockIdx.x;
  const float NEG = -3.402823466e38f;

  if (threadIdx.x < S_ / 32) msk[threadIdx.x] = (threadIdx.x == 0) ? 1u : 0u;
  __syncthreads();

  for (int st = 0; st < steps; ++st) {
    for (int i = threadIdx.x; i < S_; i += 256) sc[i] = 0.f;
    __syncthreads();

    for (int h = 0; h < H_; ++h) {
      const float* lg = logits + ((size_t)b * H_ + h) * S_;
      float mx = NEG;
      for (int i = threadIdx.x; i < S_; i += 256) {
        bool m = (msk[i >> 5] >> (i & 31)) & 1u;
        mx = fmaxf(mx, m ? NEG : lg[i]);
      }
      mx = block_reduce(mx, wred, true);
      float sum = 0.f;
      for (int i = threadIdx.x; i < S_; i += 256) {
        bool m = (msk[i >> 5] >> (i & 31)) & 1u;
        sum += m ? 0.f : __expf(lg[i] - mx);
      }
      sum = block_reduce(sum, wred, false);
      float inv = 1.0f / (sum * (float)H_);
      for (int i = threadIdx.x; i < S_; i += 256) {
        bool m = (msk[i >> 5] >> (i & 31)) & 1u;
        if (!m) sc[i] += __expf(lg[i] - mx) * inv;
      }
      __syncthreads();
    }

    float mx = NEG;
    for (int i = threadIdx.x; i < S_; i += 256) {
      bool m = (msk[i >> 5] >> (i & 31)) & 1u;
      mx = fmaxf(mx, m ? NEG : sc[i]);
    }
    mx = block_reduce(mx, wred, true);
    float sum = 0.f;
    for (int i = threadIdx.x; i < S_; i += 256) {
      bool m = (msk[i >> 5] >> (i & 31)) & 1u;
      sum += m ? 0.f : __expf(sc[i] - mx);
    }
    sum = block_reduce(sum, wred, false);
    float invZ = 1.0f / sum;
    for (int i = threadIdx.x; i < S_; i += 256) {
      bool m = (msk[i >> 5] >> (i & 31)) & 1u;
      float p = m ? 0.f : __expf(sc[i] - mx) * invZ;
      outProbs[((size_t)st * B_ + b) * S_ + i] = p;
      sc[i] = p;
    }
    __syncthreads();

    if (threadIdx.x == 0) {
      unsigned hsh = ((unsigned)st * 2654435761u) ^ ((unsigned)b * 40503u) ^ 0x9E3779B9u;
      hsh ^= hsh >> 16; hsh *= 0x7feb352du;
      hsh ^= hsh >> 15; hsh *= 0x846ca68bu;
      hsh ^= hsh >> 16;
      float u = (float)(hsh & 0xFFFFFFu) * (1.0f / 16777216.0f);
      float c = 0.f;
      int idx = 0;
      for (int i = 0; i < S_; ++i) {
        c += sc[i];
        idx = i;
        if (c >= u) break;
      }
      outIdx[(size_t)b * steps + st] = (float)idx;
      msk[idx >> 5] |= 1u << (idx & 31);
    }
    __syncthreads();
  }
}

// ---------------------------------------------------------------------------
// Host launcher
// ---------------------------------------------------------------------------
extern "C" void kernel_launch(void* const* d_in, const int* in_sizes, int n_in,
                              void* d_out, int out_size, void* d_ws, size_t ws_size,
                              hipStream_t stream)
{
  (void)in_sizes; (void)n_in; (void)ws_size;

  const float* src   = (const float*)d_in[0];
  // d_in[1] = tgt (int): only determines num_steps = tgt.shape[1] = 64
  const float* Wqkv  = (const float*)d_in[2];
  const float* bqkv  = (const float*)d_in[3];
  const float* Wo    = (const float*)d_in[4];
  const float* bo    = (const float*)d_in[5];
  const float* ln1g  = (const float*)d_in[6];
  const float* ln1b  = (const float*)d_in[7];
  const float* W1    = (const float*)d_in[8];
  const float* b1    = (const float*)d_in[9];
  const float* W2    = (const float*)d_in[10];
  const float* b2    = (const float*)d_in[11];
  const float* ln2g  = (const float*)d_in[12];
  const float* ln2b  = (const float*)d_in[13];
  const float* dWqkv = (const float*)d_in[14];
  const float* dbqkv = (const float*)d_in[15];

  const int N = NTOK;
  float* ws   = (float*)d_ws;
  float* x    = ws;                              // N*D
  float* qkvb = x + (size_t)N * D_;              // N*3D
  float* tmp  = qkvb + (size_t)N * 3 * D_;       // N*DFF (two N*D halves)
  float* tmpB = tmp + (size_t)N * D_;
  float* qdec = tmp + (size_t)N * DFF_;          // 64*D
  float* lg   = qdec + (size_t)64 * D_;          // B*H*S

  const int steps = out_size / (B_ * (1 + S_));  // = 64
  float* outIdx   = (float*)d_out;
  float* outProbs = (float*)d_out + (size_t)B_ * steps;

  dim3 blk(256);

#if USE_TDM
  auto gemm = gemm_tdm_kernel;
#else
  auto gemm = gemm_bf16_kernel;
#endif

  posenc_kernel<<<(N * D_ + 255) / 256, blk, 0, stream>>>(src, x);

  for (int l = 0; l < L_; ++l) {
    const float* Wqkv_l = Wqkv + (size_t)l * 3 * D_ * D_;
    const float* Wo_l   = Wo   + (size_t)l * D_ * D_;
    const float* W1_l   = W1   + (size_t)l * DFF_ * D_;
    const float* W2_l   = W2   + (size_t)l * D_ * DFF_;

    gemm<<<dim3(N / 64, 3 * D_ / 64), blk, 0, stream>>>(
        x, Wqkv_l, bqkv + l * 3 * D_, nullptr, qkvb, N, D_, 3 * D_, 0);
    flash_attn_kernel<<<dim3(B_ * H_ * (S_ / 128)), blk, 0, stream>>>(qkvb, tmp);
    gemm<<<dim3(N / 64, D_ / 64), blk, 0, stream>>>(
        tmp, Wo_l, bo + l * D_, x, tmpB, N, D_, D_, 0);
    layernorm_kernel<<<N, blk, 0, stream>>>(tmpB, ln1g + l * D_, ln1b + l * D_, x);
    gemm<<<dim3(N / 64, DFF_ / 64), blk, 0, stream>>>(
        x, W1_l, b1 + l * DFF_, nullptr, tmp, N, D_, DFF_, 1);
    gemm<<<dim3(N / 64, D_ / 64), blk, 0, stream>>>(
        tmp, W2_l, b2 + l * D_, x, qkvb, N, DFF_, D_, 0);
    layernorm_kernel<<<N, blk, 0, stream>>>(qkvb, ln2g + l * D_, ln2b + l * D_, x);
  }

  gemm<<<dim3(1, D_ / 64), blk, 0, stream>>>(
      x, dWqkv, dbqkv, nullptr, qdec, 64, D_, D_, 0);
  gemm<<<dim3(N / 64, D_ / 64), blk, 0, stream>>>(
      x, dWqkv + (size_t)D_ * D_, dbqkv + D_, nullptr, qkvb, N, D_, D_, 0);

  dec_logits_kernel<<<(B_ * H_ * S_ + 255) / 256, blk, 0, stream>>>(qdec, qkvb, lg);

  sampler_kernel<<<B_, blk, 0, stream>>>(lg, outIdx, outProbs, steps);
}
